// MultisourcesWindowedCrossAttention_43800076485203
// MI455X (gfx1250) — matrix operand order repaired and checked by reading, not compiled, not run on hardware
//
#include <hip/hip_runtime.h>
#include <hip/hip_bf16.h>

typedef _Float16 v16h __attribute__((ext_vector_type(16)));
typedef _Float16 v8h  __attribute__((ext_vector_type(8)));
typedef float    v8f  __attribute__((ext_vector_type(8)));
typedef float    v4f  __attribute__((ext_vector_type(4)));

#define WMMA(a, b, c) __builtin_amdgcn_wmma_f32_16x16x32_f16( \
    false, (a), false, (b), (short)0, (c), false, false)

// ---- problem constants ----
static constexpr int BB    = 2;     // batch
static constexpr int HWDIM = 224;   // image H=W
static constexpr int DV    = 256;   // value channels
static constexpr int DC    = 64;    // coord channels
static constexpr int NWIN  = 28;    // windows per image dim
static constexpr int NPS   = 784;   // windows per source
static constexpr int NTOT  = 1568;  // windows total (2 sources)
static constexpr int TOK   = 64;    // tokens per window (8x8)
static constexpr int EC    = 32;    // packed q/k dim (16 + 4 + 12 pad) == WMMA K
static constexpr int VD    = 512;   // per-head value dim (64 tok * 8)
static constexpr int NHD   = 8;
static constexpr size_t SRC_ELEMS = (size_t)BB * HWDIM * HWDIM * DV; // one output tensor

static constexpr float INV_S128 = 0.08838834764831845f;  // 1/sqrt(128)
static constexpr float INV_S32  = 0.17677669529663687f;  // 1/sqrt(32)

// ---- workspace layout (bytes) ----
static constexpr size_t OFF_VVT   = 0;                                   // f16 [16][512][1568]
static constexpr size_t SZ_VVT    = (size_t)16 * VD * NTOT * 2;
static constexpr size_t OFF_QCAT  = OFF_VVT + SZ_VVT;                    // f16 [16][1568][32]
static constexpr size_t SZ_QCAT   = (size_t)16 * NTOT * EC * 2;
static constexpr size_t OFF_KCAT  = OFF_QCAT + SZ_QCAT;
static constexpr size_t OFF_RWP   = OFF_KCAT + SZ_QCAT;                  // f16 [2][1568][64][64]
static constexpr size_t SZ_RWP    = (size_t)BB * NTOT * TOK * 64 * 2;
static constexpr size_t OFF_WVT   = OFF_RWP + SZ_RWP;                    // f16 [64][256]
static constexpr size_t OFF_WBKT  = OFF_WVT + (size_t)64 * 256 * 2;      // f16 [256][64]

// A-fragment (16x32 f16): lane holds row M=lane%16; 8 halfs at +0 and 8 at +16
// relative to the passed pointer (caller already added the h0 = 8*(lane>=16) offset).
__device__ inline v16h ldA(const _Float16* p) {
    v8h lo = *(const v8h*)p;
    v8h hi = *(const v8h*)(p + 16);
    v16h r;
#pragma unroll
    for (int i = 0; i < 8; ++i) { r[i] = lo[i]; r[8 + i] = hi[i]; }
    return r;
}

// B-fragment (32x16 f16): lane holds col N=lane%16; 8 halfs at +0 and 8 at +16
// (K-major source, caller already added h0 = 8*(lane>=16)).
__device__ inline v16h ldB(const _Float16* p) {
    v8h lo = *(const v8h*)p;
    v8h hi = *(const v8h*)(p + 16);
    v16h r;
#pragma unroll
    for (int i = 0; i < 8; ++i) { r[i] = lo[i]; r[8 + i] = hi[i]; }
    return r;
}

// ---------------------------------------------------------------------------
// prep: Wv (256x64 f32) -> WvT f16 [64][256]; Wback (64x256 f32) -> WbackT f16 [256][64]
__global__ void prep_kernel(const float* __restrict__ Wv, const float* __restrict__ Wback,
                            _Float16* __restrict__ WvT, _Float16* __restrict__ WbackT) {
    int i = blockIdx.x * 256 + threadIdx.x;  // 0..16383
    {   int n = i >> 8, k = i & 255;  WvT[i]    = (_Float16)Wv[k * 64 + n];   }
    {   int n = i >> 6, k = i & 63;   WbackT[i] = (_Float16)Wback[k * 256 + n]; }
}

// ---------------------------------------------------------------------------
// qk: 8 windows per block (one wave per window for the rms reduction).
// Per block: window means -> qk projection (Wqkv read ONCE per block, 8 FMA/load)
// -> rmsnorm -> packed Q/K f16 with score scales folded into Q.
__global__ __launch_bounds__(256, 1)
void qk_kernel(const float* __restrict__ v0, const float* __restrict__ c0,
               const float* __restrict__ v1, const float* __restrict__ c1,
               const float* __restrict__ Wqkv, const float* __restrict__ bqkv,
               const float* __restrict__ gv,
               const float* __restrict__ Wqkc, const float* __restrict__ bqkc,
               const float* __restrict__ gc,
               _Float16* __restrict__ Qcat, _Float16* __restrict__ Kcat) {
    constexpr int WBL = 8;  // windows per block == waves per block
    __shared__ float vavg[WBL][256], qv[WBL][256];
    __shared__ float cavg[WBL][64], qc[WBL][64];
    __shared__ float scal[WBL][2];
    const int tid = threadIdx.x;
    const int l0  = blockIdx.x * WBL;   // linear (b,window) base, 0..3135

    // --- window means ---
    for (int wi = 0; wi < WBL; ++wi) {
        const int l = l0 + wi;
        const int b = l / NTOT, wg = l % NTOT;
        const int src = wg / NPS, wloc = wg % NPS;
        const int wh = wloc / NWIN, ww = wloc % NWIN;
        const float* V = src ? v1 : v0;
        const float* C = src ? c1 : c0;
        const size_t pixbase = ((size_t)b * HWDIM + wh * 8) * HWDIM + ww * 8;
        float s = 0.f;
#pragma unroll 8
        for (int t = 0; t < 64; ++t) {
            size_t p = pixbase + (size_t)(t >> 3) * HWDIM + (t & 7);
            s += V[p * DV + tid];
        }
        vavg[wi][tid] = s * (1.f / 64.f);
        if (tid < 64) {
            float sc = 0.f;
#pragma unroll 8
            for (int t = 0; t < 64; ++t) {
                size_t p = pixbase + (size_t)(t >> 3) * HWDIM + (t & 7);
                sc += C[p * DC + tid];
            }
            cavg[wi][tid] = sc * (1.f / 64.f);
        }
    }
    __syncthreads();

    // --- projections: one pass over weights serves all 8 windows ---
    {
        float acc[WBL];
        const float bv = bqkv[tid];
#pragma unroll
        for (int wi = 0; wi < WBL; ++wi) acc[wi] = bv;
        for (int i = 0; i < 256; ++i) {
            const float wgt = Wqkv[i * 256 + tid];
#pragma unroll
            for (int wi = 0; wi < WBL; ++wi) acc[wi] += vavg[wi][i] * wgt;
        }
#pragma unroll
        for (int wi = 0; wi < WBL; ++wi) qv[wi][tid] = acc[wi];
        if (tid < 64) {
            float accc[WBL];
            const float bc = bqkc[tid];
#pragma unroll
            for (int wi = 0; wi < WBL; ++wi) accc[wi] = bc;
            for (int i = 0; i < 64; ++i) {
                const float wgt = Wqkc[i * 64 + tid];
#pragma unroll
                for (int wi = 0; wi < WBL; ++wi) accc[wi] += cavg[wi][i] * wgt;
            }
#pragma unroll
            for (int wi = 0; wi < WBL; ++wi) qc[wi][tid] = accc[wi];
        }
    }
    __syncthreads();

    // --- rms scales: wave w reduces window w via shuffles ---
    {
        const int wi = tid >> 5, ln = tid & 31;
        float s = 0.f;
#pragma unroll
        for (int j = 0; j < 8; ++j) { float x = qv[wi][ln + 32 * j]; s += x * x; }
        float sc = 0.f;
#pragma unroll
        for (int j = 0; j < 2; ++j) { float x = qc[wi][ln + 32 * j]; sc += x * x; }
#pragma unroll
        for (int off = 16; off > 0; off >>= 1) {
            s  += __shfl_xor(s,  off, 32);
            sc += __shfl_xor(sc, off, 32);
        }
        if (ln == 0) {
            scal[wi][0] = rsqrtf(s  * (1.f / 256.f) + 1e-6f);
            scal[wi][1] = rsqrtf(sc * (1.f / 64.f)  + 1e-6f);
        }
    }
    __syncthreads();

    // --- normalize in place ---
    {
        const float g256 = gv[tid];
        const float g64  = (tid < 64) ? gc[tid] : 0.f;
#pragma unroll
        for (int wi = 0; wi < WBL; ++wi) {
            qv[wi][tid] *= scal[wi][0] * g256;
            if (tid < 64) qc[wi][tid] *= scal[wi][1] * g64;
        }
    }
    __syncthreads();

    // --- pack per head: q = [qv(16)/sqrt128 | qc(4)/sqrt32 | 0], k = [kv | kc | 0] ---
    const int hd = tid >> 5, e = tid & 31;
#pragma unroll
    for (int wi = 0; wi < WBL; ++wi) {
        const int l = l0 + wi;
        const int b = l / NTOT, wg = l % NTOT;
        const size_t o = ((size_t)(b * NHD + hd) * NTOT + wg) * EC + e;
        float qval = 0.f, kval = 0.f;
        if (e < 16)      { qval = qv[wi][hd * 32 + e] * INV_S128;      kval = qv[wi][hd * 32 + 16 + e]; }
        else if (e < 20) { qval = qc[wi][hd * 8 + (e - 16)] * INV_S32; kval = qc[wi][hd * 8 + 4 + (e - 16)]; }
        Qcat[o] = (_Float16)qval;
        Kcat[o] = (_Float16)kval;
    }
}

// ---------------------------------------------------------------------------
// proj: tokens x 256 @ Wv(256x64) via WMMA -> VvT f16 [b*8+h][tok*8+ei][win]
__global__ __launch_bounds__(128, 1)
void proj_kernel(const float* __restrict__ v0, const float* __restrict__ v1,
                 const _Float16* __restrict__ WvTg, _Float16* __restrict__ VvT) {
    __shared__ _Float16 wv[64 * 256];
    for (int i = threadIdx.x; i < 64 * 256; i += 128) wv[i] = WvTg[i];
    __syncthreads();

    const int lane = threadIdx.x & 31, wid = threadIdx.x >> 5;
    const int nn = lane & 15, hf = lane >> 4;
    const int h0 = hf * 8, klow = hf * 16;
    const int tile = blockIdx.x * 4 + wid;      // 0..12543
    const int tr = tile * 16;                   // first global token row
    const int b    = tr / (NTOT * TOK);
    const int rem  = tr % (NTOT * TOK);
    const int win  = rem / TOK;                 // same window for all 16 rows
    const int tb   = rem & 63;
    const int src  = win / NPS, wloc = win % NPS;
    const int wh = wloc / NWIN, ww = wloc % NWIN;
    const float* V = src ? v1 : v0;
    const int tok  = tb + nn;                   // this lane's A-row token
    const size_t pix = ((size_t)b * HWDIM + wh * 8 + (tok >> 3)) * HWDIM + ww * 8 + (tok & 7);
    const float* vp = V + pix * DV;

    const v8f z = {0.f, 0.f, 0.f, 0.f, 0.f, 0.f, 0.f, 0.f};
    v8f acc[4] = {z, z, z, z};
#pragma unroll
    for (int ks = 0; ks < 8; ++ks) {
        v4f f0 = *(const v4f*)(vp + ks * 32 + h0);
        v4f f1 = *(const v4f*)(vp + ks * 32 + h0 + 4);
        v4f f2 = *(const v4f*)(vp + ks * 32 + h0 + 16);
        v4f f3 = *(const v4f*)(vp + ks * 32 + h0 + 20);
        v16h a;
#pragma unroll
        for (int i = 0; i < 4; ++i) {
            a[i]      = (_Float16)f0[i];
            a[4 + i]  = (_Float16)f1[i];
            a[8 + i]  = (_Float16)f2[i];
            a[12 + i] = (_Float16)f3[i];
        }
#pragma unroll
        for (int ct = 0; ct < 4; ++ct) {
            v16h bf = ldB(&wv[(ct * 16 + nn) * 256 + ks * 32 + h0]);
            acc[ct] = WMMA(a, bf, acc[ct]);
        }
    }
#pragma unroll
    for (int ct = 0; ct < 4; ++ct) {
        const int ch = ct * 16 + nn, hd = ch & 7, ei = ch >> 3;
#pragma unroll
        for (int v = 0; v < 8; ++v) {
            const int tk = tb + v + h0;  // C layout: row M = v + 8*hf
            VvT[((size_t)(b * NHD + hd) * VD + tk * 8 + ei) * NTOT + win] = (_Float16)acc[ct][v];
        }
    }
}

// ---------------------------------------------------------------------------
// attn: per (b,h) flash attention, 32 q-windows per block, 8 waves own 64-wide V slices.
__global__ __launch_bounds__(256, 1)
void attn_kernel(const _Float16* __restrict__ Qcat, const _Float16* __restrict__ Kcat,
                 const _Float16* __restrict__ VvT, _Float16* __restrict__ rwp) {
    __shared__ _Float16 pbuf[8][32][48];   // per-wave private P staging (f32 C -> f16 A)
    const int lane = threadIdx.x & 31, w = threadIdx.x >> 5;
    const int qt = blockIdx.x % 49;
    const int bh = blockIdx.x / 49;
    const int b = bh >> 3, h = bh & 7;
    const int nn = lane & 15, hf = lane >> 4;
    const int h0 = hf * 8, klow = hf * 16;
    const int q0 = qt * 32;

    v16h aq[2];
#pragma unroll
    for (int rt = 0; rt < 2; ++rt)
        aq[rt] = ldA(Qcat + ((size_t)bh * NTOT + (q0 + rt * 16 + nn)) * EC + h0);

    const v8f z = {0.f, 0.f, 0.f, 0.f, 0.f, 0.f, 0.f, 0.f};
    v8f acc[2][4] = {{z, z, z, z}, {z, z, z, z}};
    float lsum[2][8] = {};
    const _Float16* vbase = VvT + (size_t)bh * VD * NTOT;
    const _Float16* kbase_p = Kcat + (size_t)bh * NTOT * EC;

    for (int kt = 0; kt < 49; ++kt) {
        const int kb = kt * 32;
        // prefetch next key-step's K/V cachelines while this step computes
        if (kt + 1 < 49) {
            const int kn = kb + 32;
            __builtin_prefetch(kbase_p + (size_t)(kn + nn) * EC + klow, 0, 3);
            __builtin_prefetch(kbase_p + (size_t)(kn + 16 + nn) * EC + klow, 0, 3);
#pragma unroll
            for (int ct = 0; ct < 4; ++ct) {
                const int n = w * 64 + ct * 16 + nn;
                __builtin_prefetch(vbase + (size_t)n * NTOT + kn + klow, 0, 3);
            }
        }
        v16h bk0 = ldB(kbase_p + (size_t)(kb + nn) * EC + h0);
        v16h bk1 = ldB(kbase_p + (size_t)(kb + 16 + nn) * EC + h0);
        v8f s[2][2];
        s[0][0] = WMMA(aq[0], bk0, z);
        s[0][1] = WMMA(aq[0], bk1, z);
        s[1][0] = WMMA(aq[1], bk0, z);
        s[1][1] = WMMA(aq[1], bk1, z);
        // exp (scores are rmsnorm-bounded; max-free softmax), accumulate row sums, stage P
#pragma unroll
        for (int rt = 0; rt < 2; ++rt)
#pragma unroll
            for (int c2 = 0; c2 < 2; ++c2)
#pragma unroll
                for (int v = 0; v < 8; ++v) {
                    float p = __expf(s[rt][c2][v]);
                    lsum[rt][v] += p;
                    pbuf[w][rt * 16 + v + h0][c2 * 16 + nn] = (_Float16)p;
                }
        // re-read P as f16 A fragments (same-wave DS ops are in-order)
        v16h ap0 = ldA(&pbuf[w][nn][h0]);
        v16h ap1 = ldA(&pbuf[w][16 + nn][h0]);
#pragma unroll
        for (int ct = 0; ct < 4; ++ct) {
            const int n = w * 64 + ct * 16 + nn;
            v16h bv = ldB(vbase + (size_t)n * NTOT + kb + h0);
            acc[0][ct] = WMMA(ap0, bv, acc[0][ct]);
            acc[1][ct] = WMMA(ap1, bv, acc[1][ct]);
        }
    }
    // finalize: full row sums across the 16-lane half, normalize, scatter f16
#pragma unroll
    for (int rt = 0; rt < 2; ++rt) {
        float inv[8];
#pragma unroll
        for (int v = 0; v < 8; ++v) {
            float s = lsum[rt][v];
            s += __shfl_xor(s, 1, 32);
            s += __shfl_xor(s, 2, 32);
            s += __shfl_xor(s, 4, 32);
            s += __shfl_xor(s, 8, 32);
            inv[v] = 1.0f / s;
        }
#pragma unroll
        for (int ct = 0; ct < 4; ++ct) {
            const int col = w * 64 + ct * 16 + nn;
            const int tok = col >> 3, ei = col & 7;
#pragma unroll
            for (int v = 0; v < 8; ++v) {
                const int q = q0 + rt * 16 + v + h0;
                rwp[(((size_t)b * NTOT + q) * TOK + tok) * 64 + ei * 8 + h] =
                    (_Float16)(acc[rt][ct][v] * inv[v]);
            }
        }
    }
}

// ---------------------------------------------------------------------------
// back: tokens x 64 @ Wback(64x256) via WMMA, scatter f32 into un-windowed outputs.
__global__ __launch_bounds__(128, 1)
void back_kernel(const _Float16* __restrict__ rwp, const _Float16* __restrict__ WbackTg,
                 float* __restrict__ out) {
    __shared__ _Float16 wb[256 * 64];
    for (int i = threadIdx.x; i < 256 * 64; i += 128) wb[i] = WbackTg[i];
    __syncthreads();

    const int lane = threadIdx.x & 31, wid = threadIdx.x >> 5;
    const int nn = lane & 15, hf = lane >> 4;
    const int h0 = hf * 8, klow = hf * 16;
    const int tile = blockIdx.x * 4 + wid;
    const int tr = tile * 16;
    const int b   = tr / (NTOT * TOK);
    const int rem = tr % (NTOT * TOK);
    const int win = rem / TOK;
    const int tb  = rem & 63;
    const int src = win / NPS, wloc = win % NPS;
    const int wh = wloc / NWIN, ww = wloc % NWIN;
    float* op = out + (size_t)src * SRC_ELEMS;

    const _Float16* ar = rwp + (size_t)(tr + nn) * 64;
    v16h a0 = ldA(ar + h0);
    v16h a1 = ldA(ar + 32 + h0);
    const v8f z = {0.f, 0.f, 0.f, 0.f, 0.f, 0.f, 0.f, 0.f};
#pragma unroll
    for (int ct = 0; ct < 16; ++ct) {
        const int n = ct * 16 + nn;
        v16h b0 = ldB(&wb[n * 64 + h0]);
        v16h b1 = ldB(&wb[n * 64 + 32 + h0]);
        v8f c = z;
        c = WMMA(a0, b0, c);
        c = WMMA(a1, b1, c);
#pragma unroll
        for (int v = 0; v < 8; ++v) {
            const int tk = tb + v + h0;
            const int hh = wh * 8 + (tk >> 3), wp = ww * 8 + (tk & 7);
            op[(((size_t)b * HWDIM + hh) * HWDIM + wp) * DV + n] = c[v];
        }
    }
}

// ---------------------------------------------------------------------------
extern "C" void kernel_launch(void* const* d_in, const int* in_sizes, int n_in,
                              void* d_out, int out_size, void* d_ws, size_t ws_size,
                              hipStream_t stream) {
    const float* values0 = (const float*)d_in[0];
    const float* coords0 = (const float*)d_in[1];
    const float* values1 = (const float*)d_in[2];
    const float* coords1 = (const float*)d_in[3];
    const float* Wqk_v   = (const float*)d_in[4];
    const float* bqk_v   = (const float*)d_in[5];
    const float* g_v     = (const float*)d_in[6];
    const float* Wqk_c   = (const float*)d_in[7];
    const float* bqk_c   = (const float*)d_in[8];
    const float* g_c     = (const float*)d_in[9];
    const float* Wv      = (const float*)d_in[10];
    const float* Wback   = (const float*)d_in[11];

    char* ws = (char*)d_ws;
    _Float16* VvT    = (_Float16*)(ws + OFF_VVT);
    _Float16* Qcat   = (_Float16*)(ws + OFF_QCAT);
    _Float16* Kcat   = (_Float16*)(ws + OFF_KCAT);
    _Float16* rwp    = (_Float16*)(ws + OFF_RWP);
    _Float16* WvT    = (_Float16*)(ws + OFF_WVT);
    _Float16* WbackT = (_Float16*)(ws + OFF_WBKT);
    float* out = (float*)d_out;

    prep_kernel<<<64, 256, 0, stream>>>(Wv, Wback, WvT, WbackT);
    qk_kernel<<<392, 256, 0, stream>>>(values0, coords0, values1, coords1,
                                       Wqk_v, bqk_v, g_v, Wqk_c, bqk_c, g_c,
                                       Qcat, Kcat);
    proj_kernel<<<3136, 128, 0, stream>>>(values0, values1, WvT, VvT);
    attn_kernel<<<16 * 49, 256, 0, stream>>>(Qcat, Kcat, VvT, rwp);
    back_kernel<<<3136, 128, 0, stream>>>(rwp, WbackT, out);
}
